// CNN_GAT_FREQ_50646254355069
// MI455X (gfx1250) — matrix operand order repaired and checked by reading, not compiled, not run on hardware
//
#include <hip/hip_runtime.h>
#include <hip/hip_bf16.h>

typedef _Float16 f16;
typedef __attribute__((ext_vector_type(8)))  _Float16 v8h;
typedef __attribute__((ext_vector_type(16))) _Float16 v16h;
typedef __attribute__((ext_vector_type(8)))  float    v8f;

#define BATCH 128
#define CIN   17
#define TLEN  8192
#define OC1   32
#define T1    4096
#define OC2   64
#define T2    2048
#define HF    16
#define EPSV  1e-5f

__device__ __forceinline__ v16h cat8(v8h a, v8h b) {
    return __builtin_shufflevector(a, b, 0,1,2,3,4,5,6,7,8,9,10,11,12,13,14,15);
}

// ---------------------------------------------------------------------------
// Kernel 0: zero the GAT accumulator
// ---------------------------------------------------------------------------
__global__ void zero_kernel(float* __restrict__ p, int n) {
    int i = blockIdx.x * blockDim.x + threadIdx.x;
    if (i < n) p[i] = 0.0f;
}

// ---------------------------------------------------------------------------
// Kernel 1: conv1 (k=5,pad=2) + BN + ReLU + MaxPool(2) via WMMA im2col GEMM
//   A (im2col) staged row-major in LDS; B staged transposed (Bt[oc][k]).
//   Fragments load as 2x ds_load_b128 each.  Output h1: [B][T1][OC1] f16
// ---------------------------------------------------------------------------
#define A1_W 13   // row stride in v8h (104 halves; 52 dwords -> conflict-friendly)
__global__ __launch_bounds__(128) void conv1_kernel(
    const float* __restrict__ x, const float* __restrict__ w,
    const float* __restrict__ cb, const float* __restrict__ g,
    const float* __restrict__ be, const float* __restrict__ bm,
    const float* __restrict__ bv, f16* __restrict__ h1)
{
    __shared__ v8h  Ald[64][A1_W];     // im2col rows, k = c*5+tap, zero-padded to 96
    __shared__ v8h  Btld[32][A1_W];    // Bt[oc][k]
    __shared__ float sc[OC1], sh[OC1];

    const int bb = blockIdx.y;
    const int t0 = blockIdx.x * 64;
    const int tid = threadIdx.x;

    // stage A: 64 rows x 12 v8h chunks (k < 96)
    for (int task = tid; task < 64 * 12; task += 128) {
        int m = task / 12, q = task % 12;
        v8h v = {};
#pragma unroll
        for (int j = 0; j < 8; ++j) {
            int k = q * 8 + j;
            float val = 0.0f;
            if (k < 85) {
                int c = k / 5, tap = k - c * 5;
                int t = t0 + m - 2 + tap;
                if (t >= 0 && t < TLEN) val = x[((size_t)bb * CIN + c) * TLEN + t];
            }
            v[j] = (f16)val;
        }
        Ald[m][q] = v;
    }
    // stage Bt: 32 rows x 12 chunks
    for (int task = tid; task < 32 * 12; task += 128) {
        int oc = task / 12, q = task % 12;
        v8h v = {};
#pragma unroll
        for (int j = 0; j < 8; ++j) {
            int k = q * 8 + j;
            float val = 0.0f;
            if (k < 85) { int c = k / 5, tap = k - c * 5; val = w[((size_t)oc * CIN + c) * 5 + tap]; }
            v[j] = (f16)val;
        }
        Btld[oc][q] = v;
    }
    if (tid < OC1) {
        float s = g[tid] * rsqrtf(bv[tid] + EPSV);
        sc[tid] = s;
        sh[tid] = be[tid] + (cb[tid] - bm[tid]) * s;
    }
    __syncthreads();

    const int lane = tid & 31, wid = tid >> 5;
    const int mloc = lane & 15, hi = lane >> 4;
    const int m0 = wid * 16;

    v8f acc0 = {}, acc1 = {};
#pragma unroll
    for (int kb8 = 0; kb8 < 12; kb8 += 4) {            // kb = kb8*8 in halves
        v16h av = cat8(Ald[m0 + mloc][kb8 + hi],      Ald[m0 + mloc][kb8 + 2 + hi]);
        v16h b0 = cat8(Btld[mloc][kb8 + hi],      Btld[mloc][kb8 + 2 + hi]);
        v16h b1 = cat8(Btld[16 + mloc][kb8 + hi], Btld[16 + mloc][kb8 + 2 + hi]);
        acc0 = __builtin_amdgcn_wmma_f32_16x16x32_f16(false, av, false, b0, (short)0, acc0, false, false);
        acc1 = __builtin_amdgcn_wmma_f32_16x16x32_f16(false, av, false, b1, (short)0, acc1, false, false);
    }

    // BN + ReLU + MaxPool(2): rows hi*8+2i / +2i+1 are adjacent acc elements
    const int n1base = (t0 + m0) / 2 + hi * 4;
    const float s0 = sc[mloc],      o0 = sh[mloc];
    const float s1 = sc[16 + mloc], o1 = sh[16 + mloc];
#pragma unroll
    for (int i = 0; i < 4; ++i) {
        float p0 = fmaxf(fmaxf(acc0[2 * i] * s0 + o0, 0.0f),
                         fmaxf(acc0[2 * i + 1] * s0 + o0, 0.0f));
        float p1 = fmaxf(fmaxf(acc1[2 * i] * s1 + o1, 0.0f),
                         fmaxf(acc1[2 * i + 1] * s1 + o1, 0.0f));
        size_t row = (size_t)bb * T1 + (n1base + i);
        h1[row * OC1 + mloc]      = (f16)p0;
        h1[row * OC1 + 16 + mloc] = (f16)p1;
    }
}

// ---------------------------------------------------------------------------
// Kernel 2: conv2 + BN + ReLU + MaxPool(2) via WMMA, K = 160 = 5 taps * 32 ch
//   k = tap*32 + ic  ->  each 8-half im2col chunk is one global b128 from h1.
//   Output nodes: [B][T2][OC2] f16
// ---------------------------------------------------------------------------
#define A2_W 21   // row stride in v8h (168 halves; 84 dwords)
__global__ __launch_bounds__(128) void conv2_kernel(
    const f16* __restrict__ h1, const float* __restrict__ w,
    const float* __restrict__ cb, const float* __restrict__ g,
    const float* __restrict__ be, const float* __restrict__ bm,
    const float* __restrict__ bv, f16* __restrict__ nodes)
{
    __shared__ v8h  Ald[64][A2_W];     // im2col rows, K=160 (q < 20)
    __shared__ v8h  Btld[64][A2_W];    // Bt[oc][k]
    __shared__ float sc[OC2], sh[OC2];

    const int bb = blockIdx.y;
    const int t0 = blockIdx.x * 64;
    const int tid = threadIdx.x;

    // stage A: one aligned global b128 per chunk (tap = q>>2, ic base = (q&3)*8)
    for (int task = tid; task < 64 * 20; task += 128) {
        int m = task / 20, q = task % 20;
        int tap = q >> 2, icb = (q & 3) * 8;
        int t = t0 + m - 2 + tap;
        v8h v = {};
        if (t >= 0 && t < T1)
            v = *(const v8h*)(h1 + ((size_t)bb * T1 + t) * OC1 + icb);
        Ald[m][q] = v;
    }
    // stage Bt
    for (int task = tid; task < 64 * 20; task += 128) {
        int oc = task / 20, q = task % 20;
        v8h v = {};
#pragma unroll
        for (int j = 0; j < 8; ++j) {
            int k = q * 8 + j;
            int ic = k & 31, tap = k >> 5;
            v[j] = (f16)w[((size_t)oc * OC1 + ic) * 5 + tap];
        }
        Btld[oc][q] = v;
    }
    for (int c = tid; c < OC2; c += 128) {
        float s = g[c] * rsqrtf(bv[c] + EPSV);
        sc[c] = s;
        sh[c] = be[c] + (cb[c] - bm[c]) * s;
    }
    __syncthreads();

    const int lane = tid & 31, wid = tid >> 5;
    const int mloc = lane & 15, hi = lane >> 4;
    const int m0 = wid * 16;

    v8f acc[4] = {v8f{}, v8f{}, v8f{}, v8f{}};
#pragma unroll
    for (int kb8 = 0; kb8 < 20; kb8 += 4) {
        v16h av = cat8(Ald[m0 + mloc][kb8 + hi], Ald[m0 + mloc][kb8 + 2 + hi]);
#pragma unroll
        for (int j = 0; j < 4; ++j) {
            v16h bvv = cat8(Btld[j * 16 + mloc][kb8 + hi],
                            Btld[j * 16 + mloc][kb8 + 2 + hi]);
            acc[j] = __builtin_amdgcn_wmma_f32_16x16x32_f16(false, av, false, bvv, (short)0, acc[j], false, false);
        }
    }

    const int n2base = (t0 + m0) / 2 + hi * 4;
#pragma unroll
    for (int j = 0; j < 4; ++j) {
        int ch = j * 16 + mloc;
        float s = sc[ch], o = sh[ch];
#pragma unroll
        for (int i = 0; i < 4; ++i) {
            float p = fmaxf(fmaxf(acc[j][2 * i] * s + o, 0.0f),
                            fmaxf(acc[j][2 * i + 1] * s + o, 0.0f));
            nodes[((size_t)bb * T2 + (n2base + i)) * OC2 + ch] = (f16)p;
        }
    }
}

// ---------------------------------------------------------------------------
// Kernel 3: wx = nodes @ gat_w  ([2048 x 64] x [64 x 16]) per batch, WMMA
//   A fragments come straight from global nodes rows as b128 loads.
//   Output wx: [B][T2][16] f32
// ---------------------------------------------------------------------------
#define GW_W 9    // 72 halves stride (36 dwords)
__global__ __launch_bounds__(128) void wx_kernel(
    const f16* __restrict__ nodes, const float* __restrict__ gw,
    float* __restrict__ wx)
{
    __shared__ v8h Gt[HF][GW_W];       // Bt[n][k], q < 8 used
    const int bb = blockIdx.y;
    const int node0 = blockIdx.x * 64;
    const int tid = threadIdx.x;

    {   // 16*8 = 128 chunks: exactly one per thread
        int n = tid >> 3, q = tid & 7;
        v8h v;
#pragma unroll
        for (int j = 0; j < 8; ++j) {
            int k = q * 8 + j;
            v[j] = (f16)gw[k * HF + n];
        }
        Gt[n][q] = v;
    }
    __syncthreads();

    const int lane = tid & 31, wid = tid >> 5;
    const int mloc = lane & 15, hi = lane >> 4;
    const int m0 = wid * 16;
    const v8h* arow = (const v8h*)(nodes + ((size_t)bb * T2 + node0 + m0 + mloc) * OC2);

    v8f acc = {};
#pragma unroll
    for (int kb8 = 0; kb8 < 8; kb8 += 4) {
        v16h av  = cat8(arow[kb8 + hi], arow[kb8 + 2 + hi]);
        v16h bvv = cat8(Gt[mloc][kb8 + hi], Gt[mloc][kb8 + 2 + hi]);
        acc = __builtin_amdgcn_wmma_f32_16x16x32_f16(false, av, false, bvv, (short)0, acc, false, false);
    }
#pragma unroll
    for (int r = 0; r < 8; ++r) {
        int node = node0 + m0 + hi * 8 + r;
        wx[((size_t)bb * T2 + node) * HF + mloc] = acc[r];
    }
}

// ---------------------------------------------------------------------------
// Kernel 4: banded (+/-2) GAT softmax + aggregation + partial mean
// ---------------------------------------------------------------------------
__global__ __launch_bounds__(256) void gat_agg_kernel(
    const float* __restrict__ wx, const float* __restrict__ asrc,
    const float* __restrict__ adst, float* __restrict__ accum)
{
    __shared__ float red[256][HF];
    const int bb = blockIdx.y;
    const int n = blockIdx.x * 256 + threadIdx.x;
    const float* wxb = wx + (size_t)bb * T2 * HF;

    float e_dst = 0.0f;
#pragma unroll
    for (int h = 0; h < HF; ++h) e_dst += wxb[(size_t)n * HF + h] * adst[h];

    float logit[5];
    float mx = -3.0e38f;
#pragma unroll
    for (int d = 0; d < 5; ++d) {
        int p = n + d - 2;
        float l = -1.0e9f;
        if (p >= 0 && p < T2) {
            float e = e_dst;
#pragma unroll
            for (int h = 0; h < HF; ++h) e += wxb[(size_t)p * HF + h] * asrc[h];
            l = (e > 0.0f) ? e : 0.2f * e;   // leaky relu
        }
        logit[d] = l;
        mx = fmaxf(mx, l);
    }
    float al[5], ssum = 0.0f;
#pragma unroll
    for (int d = 0; d < 5; ++d) {
        int p = n + d - 2;
        float ev = (p >= 0 && p < T2) ? __expf(logit[d] - mx) : 0.0f;
        al[d] = ev;
        ssum += ev;
    }
    float inv = 1.0f / ssum;

    float outh[HF];
#pragma unroll
    for (int h = 0; h < HF; ++h) outh[h] = 0.0f;
#pragma unroll
    for (int d = 0; d < 5; ++d) {
        int p = n + d - 2;
        if (p >= 0 && p < T2) {
            float a = al[d] * inv;
#pragma unroll
            for (int h = 0; h < HF; ++h) outh[h] += a * wxb[(size_t)p * HF + h];
        }
    }

#pragma unroll
    for (int h = 0; h < HF; ++h) red[threadIdx.x][h] = outh[h];
    __syncthreads();
    for (int s = 128; s >= 1; s >>= 1) {
        if ((int)threadIdx.x < s) {
#pragma unroll
            for (int h = 0; h < HF; ++h) red[threadIdx.x][h] += red[threadIdx.x + s][h];
        }
        __syncthreads();
    }
    if (threadIdx.x < HF)
        atomicAdd(&accum[bb * HF + threadIdx.x], red[0][threadIdx.x]);
}

// ---------------------------------------------------------------------------
// Kernel 5: freq head + concat + fc  (one thread per batch row)
// ---------------------------------------------------------------------------
__global__ __launch_bounds__(128) void final_kernel(
    const float* __restrict__ accum, const float* __restrict__ gbias,
    const float* __restrict__ delta, const float* __restrict__ theta,
    const float* __restrict__ fw, const float* __restrict__ fb,
    const float* __restrict__ fcw, const float* __restrict__ fcb,
    float* __restrict__ out)
{
    int bb = threadIdx.x;
    if (bb >= BATCH) return;
    float comb[50];
#pragma unroll
    for (int h = 0; h < HF; ++h)
        comb[h] = accum[bb * HF + h] * (1.0f / (float)T2) + gbias[h];
    for (int j = 0; j < 34; ++j) {
        float s = fb[j];
        for (int i = 0; i < 17; ++i) s += delta[bb * 17 + i] * fw[j * 34 + i];
        for (int i = 0; i < 17; ++i) s += theta[bb * 17 + i] * fw[j * 34 + 17 + i];
        comb[HF + j] = 1.5f * fmaxf(s, 0.0f);
    }
#pragma unroll
    for (int o = 0; o < 2; ++o) {
        float s = fcb[o];
        for (int i = 0; i < 50; ++i) s += comb[i] * fcw[o * 50 + i];
        out[bb * 2 + o] = s;
    }
}

// ---------------------------------------------------------------------------
extern "C" void kernel_launch(void* const* d_in, const int* in_sizes, int n_in,
                              void* d_out, int out_size, void* d_ws, size_t ws_size,
                              hipStream_t stream) {
    const float* x       = (const float*)d_in[0];
    const float* delta   = (const float*)d_in[1];
    const float* theta   = (const float*)d_in[2];
    const float* conv1_w = (const float*)d_in[3];
    const float* conv1_b = (const float*)d_in[4];
    const float* bn1_g   = (const float*)d_in[5];
    const float* bn1_b   = (const float*)d_in[6];
    const float* bn1_m   = (const float*)d_in[7];
    const float* bn1_v   = (const float*)d_in[8];
    const float* conv2_w = (const float*)d_in[9];
    const float* conv2_b = (const float*)d_in[10];
    const float* bn2_g   = (const float*)d_in[11];
    const float* bn2_b   = (const float*)d_in[12];
    const float* bn2_m   = (const float*)d_in[13];
    const float* bn2_v   = (const float*)d_in[14];
    const float* gat_w   = (const float*)d_in[15];
    const float* gat_as  = (const float*)d_in[16];
    const float* gat_ad  = (const float*)d_in[17];
    const float* gat_bias= (const float*)d_in[18];
    const float* freq_w  = (const float*)d_in[19];
    const float* freq_b  = (const float*)d_in[20];
    const float* fc_w    = (const float*)d_in[21];
    const float* fc_b    = (const float*)d_in[22];
    float* out = (float*)d_out;

    // workspace layout
    char* ws = (char*)d_ws;
    f16*   h1    = (f16*)(ws);                                 // 128*4096*32*2 = 32 MB
    f16*   nodes = (f16*)(ws + (size_t)33554432);              // 128*2048*64*2 = 32 MB
    float* wx    = (float*)(ws + (size_t)67108864);            // 128*2048*16*4 = 16 MB
    float* accum = (float*)(ws + (size_t)83886080);            // 128*16*4      = 8 KB

    zero_kernel<<<(BATCH * HF + 255) / 256, 256, 0, stream>>>(accum, BATCH * HF);

    conv1_kernel<<<dim3(TLEN / 64, BATCH), 128, 0, stream>>>(
        x, conv1_w, conv1_b, bn1_g, bn1_b, bn1_m, bn1_v, h1);

    conv2_kernel<<<dim3(T1 / 64, BATCH), 128, 0, stream>>>(
        h1, conv2_w, conv2_b, bn2_g, bn2_b, bn2_m, bn2_v, nodes);

    wx_kernel<<<dim3(T2 / 64, BATCH), 128, 0, stream>>>(nodes, gat_w, wx);

    gat_agg_kernel<<<dim3(T2 / 256, BATCH), 256, 0, stream>>>(wx, gat_as, gat_ad, accum);

    final_kernel<<<1, 128, 0, stream>>>(accum, gat_bias, delta, theta,
                                        freq_w, freq_b, fc_w, fc_b, out);
}
